// MultiHeadedHaloAttention_12962211299917
// MI455X (gfx1250) — matrix operand order repaired and checked by reading, not compiled, not run on hardware
//
#include <hip/hip_runtime.h>
#include <hip/hip_bf16.h>

typedef __bf16 bf16;
typedef __attribute__((ext_vector_type(16))) __bf16 v16bf;
typedef __attribute__((ext_vector_type(8)))  __bf16 v8bf;
typedef __attribute__((ext_vector_type(8)))  float  v8f;
typedef __attribute__((ext_vector_type(4)))  unsigned int uint32x4;
typedef __attribute__((ext_vector_type(4)))  int  int32x4;
typedef __attribute__((ext_vector_type(8)))  int  int32x8;

typedef int vint4 __attribute__((vector_size(16)));            // b128 payload type
typedef __attribute__((address_space(1))) vint4 as1_vint4;     // global
typedef __attribute__((address_space(3))) vint4 as3_vint4;     // LDS
typedef __attribute__((address_space(3))) void  las_void;      // LDS (for offset math)

#if __has_builtin(__builtin_amdgcn_global_load_async_to_lds_b128)
#define HAVE_ASYNC 1
#else
#define HAVE_ASYNC 0
#endif
#if __has_builtin(__builtin_amdgcn_tensor_load_to_lds)
#define HAVE_TDM 1
#else
#define HAVE_TDM 0
#endif

__device__ __forceinline__ bf16 f2bf(float f) {
    unsigned u = __float_as_uint(f);
    u += 0x7FFFu + ((u >> 16) & 1u);           // round-to-nearest-even
    unsigned short s = (unsigned short)(u >> 16);
    return __builtin_bit_cast(bf16, s);
}

__device__ __forceinline__ v8f wmma_bf16f32(v16bf a, v16bf b, v8f c) {
    return __builtin_amdgcn_wmma_f32_16x16x32_bf16(
        false, a, false, b, (short)0, c, false, false);
}

union AFrag { v16bf v; v8bf h[2]; };

#if HAVE_ASYNC
__device__ __forceinline__ void async_cp16(void* lds, const void* g) {
    // GLOBAL_LOAD_ASYNC_TO_LDS_B128: global -> LDS without VGPR round-trip (ASYNCcnt)
    __builtin_amdgcn_global_load_async_to_lds_b128(
        (as1_vint4*)g, (as3_vint4*)lds, 0, 0);
}
__device__ __forceinline__ void wait_async0() {
#if __has_builtin(__builtin_amdgcn_s_wait_asynccnt)
    __builtin_amdgcn_s_wait_asynccnt(0);
#else
    asm volatile("s_wait_asynccnt 0" ::: "memory");
#endif
}
#endif

#if HAVE_TDM
// One-shot TDM load of a 64(d) x 8(x) x 8(y) bf16 tile into LDS (TENSORcnt).
__device__ __forceinline__ void tdm_load_q(const bf16* gsrc, unsigned lds_off) {
    unsigned long long ga = (unsigned long long)(const void*)gsrc;
    // D# group0: count=1 | lds_addr | global_addr[31:0] | global_addr[56:32], type=2
    uint32x4 g0 = { 1u, lds_off, (unsigned)ga,
                    (unsigned)((ga >> 32) & 0x1FFFFFFu) | (2u << 30) };
    // D# group1: data_size=2B; tensor_dim0=64, tensor_dim1=8;
    //            tile_dim0=64, tile_dim1=8, tile_dim2=8;
    //            dim0_stride=1536, dim1_stride=98304 (=64*1536)
    int32x8 g1 = { (int)0x10000,
                   (int)(64u << 16),                 // tensor_dim0[15:0]
                   (int)(8u  << 16),                 // tensor_dim1[15:0]
                   (int)(64u << 16),                 // tile_dim0
                   (int)(8u | (8u << 16)),           // tile_dim1 | tile_dim2
                   (int)1536,                        // tensor_dim0_stride[31:0]
                   (int)(0x8000u << 16),             // dim1_stride[15:0] in [31:16]
                   1 };                              // dim1_stride[47:16]
    int32x4 g2 = { 4096, 0, 0, 0 };                  // tensor_dim2 (OOB bound)
    int32x4 g3 = { 0, 0, 0, 0 };
#if defined(__clang_major__) && __clang_major__ >= 23
    int32x8 g4 = {};
    __builtin_amdgcn_tensor_load_to_lds(g0, g1, g2, g3, g4, 0);
#else
    __builtin_amdgcn_tensor_load_to_lds(g0, g1, g2, g3, 0);
#endif
}
#endif

// ---------------------------------------------------------------------------
// Kernel 1a: fp32 -> bf16 elementwise (x activation)
// ---------------------------------------------------------------------------
__global__ __launch_bounds__(256) void k_f32_to_bf16(const float* __restrict__ src,
                                                     bf16* __restrict__ dst, int n) {
    for (int i = blockIdx.x * blockDim.x + threadIdx.x; i < n; i += gridDim.x * blockDim.x)
        dst[i] = f2bf(src[i]);
}

// ---------------------------------------------------------------------------
// Kernel 1b: fp32 [rows][cols] -> bf16 [cols][rows]  (weight transpose)
// ---------------------------------------------------------------------------
__global__ __launch_bounds__(256) void k_transpose_bf16(const float* __restrict__ src,
                                                        bf16* __restrict__ dst,
                                                        int rows, int cols) {
    int i = blockIdx.x * blockDim.x + threadIdx.x;
    if (i < rows * cols) {
        int r = i / cols, c = i % cols;
        dst[(size_t)c * rows + r] = f2bf(src[i]);
    }
}

// ---------------------------------------------------------------------------
// Shared GEMM staging: 128x32 A-tile + 128x32 B-tile, async-to-LDS if available
// ---------------------------------------------------------------------------
__device__ __forceinline__ void stage_tiles(const bf16* ax, const bf16* bx,
                                            bf16* asr, bf16* bsr, bool more) {
#if HAVE_ASYNC
    async_cp16(asr,     ax);
    async_cp16(asr + 8, ax + 8);
    async_cp16(bsr,     bx);
    async_cp16(bsr + 8, bx + 8);
    if (more) { __builtin_prefetch(ax + 32, 0, 0); __builtin_prefetch(bx + 32, 0, 0); }
    wait_async0();
#else
    *(uint4*)asr       = *(const uint4*)ax;
    *(uint4*)(asr + 8) = *(const uint4*)(ax + 8);
    *(uint4*)bsr       = *(const uint4*)bx;
    *(uint4*)(bsr + 8) = *(const uint4*)(bx + 8);
    if (more) { __builtin_prefetch(ax + 32, 0, 0); __builtin_prefetch(bx + 32, 0, 0); }
#endif
}

// ---------------------------------------------------------------------------
// Kernel 2: QKV GEMM  Y[32768,1536](bf16) = X[32768,512](bf16) * WT[1536,512]^T
// ---------------------------------------------------------------------------
__global__ __launch_bounds__(256) void k_gemm_qkv(const bf16* __restrict__ X,
                                                  const bf16* __restrict__ Wt,
                                                  bf16* __restrict__ Y) {
    __shared__ __align__(32) bf16 As[128][32];
    __shared__ __align__(32) bf16 Bs[128][32];   // [n][k]

    const int m0 = blockIdx.x * 128;
    const int n0 = blockIdx.y * 128;
    const int t = threadIdx.x, lane = t & 31, wave = t >> 5;
    const int wm = (wave >> 2) * 64;
    const int wn = (wave & 3) * 32;
    const int lr = t >> 1, lseg = (t & 1) * 16;

    v8f acc[4][2] = {};
    const int khalf = (lane >> 4) * 8;
    const int kq    = (lane >> 4) * 16;
    const int ml    = lane & 15;

    for (int k0 = 0; k0 < 512; k0 += 32) {
        __syncthreads();
        stage_tiles(&X [(size_t)(m0 + lr) * 512 + k0 + lseg],
                    &Wt[(size_t)(n0 + lr) * 512 + k0 + lseg],
                    &As[lr][lseg], &Bs[lr][lseg], k0 + 32 < 512);
        __syncthreads();
        #pragma unroll
        for (int mt = 0; mt < 4; ++mt) {
            const bf16* pa = &As[wm + mt * 16 + ml][0];
            AFrag ua;
            ua.h[0] = *(const v8bf*)(pa + khalf);
            ua.h[1] = *(const v8bf*)(pa + 16 + khalf);
            #pragma unroll
            for (int nt = 0; nt < 2; ++nt) {
                AFrag ub;
                ub.h[0] = *(const v8bf*)&Bs[wn + nt * 16 + ml][khalf];
                ub.h[1] = *(const v8bf*)&Bs[wn + nt * 16 + ml][16 + khalf];
                v16bf bv = ub.v;
                acc[mt][nt] = wmma_bf16f32(ua.v, bv, acc[mt][nt]);
            }
        }
    }

    const int mrow = (lane >> 4) * 8;
    #pragma unroll
    for (int mt = 0; mt < 4; ++mt)
        #pragma unroll
        for (int nt = 0; nt < 2; ++nt)
            #pragma unroll
            for (int r = 0; r < 8; ++r) {
                size_t row = (size_t)(m0 + wm + mt * 16 + mrow + r);
                int col = n0 + wn + nt * 16 + ml;
                Y[row * 1536 + col] = f2bf(acc[mt][nt][r]);
            }
}

// ---------------------------------------------------------------------------
// Kernel 3: halo attention per (b, head, window-block).
// ---------------------------------------------------------------------------
__global__ __launch_bounds__(256) void k_halo_attn(const bf16* __restrict__ QKV,
                                                   const float* __restrict__ rel,
                                                   bf16* __restrict__ Outp) {
    __shared__ __align__(32) bf16  Qs[64][64];       //  8 KB [qpix][d]
    __shared__ __align__(32) bf16  Ks[224 * 64];     // 28 KB [key][d]; reused as P[64][224]
    __shared__ __align__(32) bf16  Vst[64][224];     // 28 KB [d][key]
    __shared__ __align__(32) float Sc[64][224];      // 56 KB [qpix][key]

    const int id = blockIdx.x;
    const int b    = id >> 9;
    const int head = (id >> 6) & 7;
    const int l    = id & 63;
    const int bh = l >> 3, bw = l & 7;
    const int t = threadIdx.x, lane = t & 31, wave = t >> 5;

    // ---- stage Q (64 rows x 64 d)
#if HAVE_TDM
    if (wave == 0) {
        const bf16* qbase = QKV + ((size_t)((b * 64 + bh * 8) * 64 + bw * 8)) * 1536 + head * 64;
        tdm_load_q(qbase, (unsigned)(unsigned long long)(las_void*)&Qs[0][0]);
        __builtin_amdgcn_s_wait_tensorcnt(0);
    }
#else
    {
        int qp = t >> 2, seg = (t & 3) * 16;
        int y = bh * 8 + (qp >> 3), x = bw * 8 + (qp & 7);
        const bf16* src = QKV + ((size_t)((b * 64 + y) * 64 + x)) * 1536 + head * 64 + seg;
        *(uint4*)&Qs[qp][seg]     = *(const uint4*)src;
        *(uint4*)&Qs[qp][seg + 8] = *(const uint4*)(src + 8);
    }
#endif
    // ---- stage K (row-major, async when available) and V (transposed), zero-pad
    if (t < 224) {
        int r = t;
        bool valid = r < 196;
        int y = 0, x = 0;
        if (valid) {
            int ky = r / 14, kx = r % 14;
            y = bh * 8 - 3 + ky; x = bw * 8 - 3 + kx;
            valid = (y >= 0) & (y < 64) & (x >= 0) & (x < 64);
        }
        bf16* krow = &Ks[r * 64];
        if (valid) {
            const bf16* ksrc = QKV + ((size_t)((b * 64 + y) * 64 + x)) * 1536 + 512 + head * 64;
            const bf16* vsrc = ksrc + 512;
#if HAVE_ASYNC
            #pragma unroll
            for (int s = 0; s < 64; s += 8) async_cp16(krow + s, ksrc + s);
#else
            #pragma unroll
            for (int s = 0; s < 64; s += 8)
                *(uint4*)(krow + s) = *(const uint4*)(ksrc + s);
#endif
            #pragma unroll
            for (int d = 0; d < 64; ++d) Vst[d][r] = vsrc[d];
        } else {
            uint4 z = make_uint4(0, 0, 0, 0);
            #pragma unroll
            for (int s = 0; s < 64; s += 8) *(uint4*)(krow + s) = z;
            bf16 zb = f2bf(0.0f);
            #pragma unroll
            for (int d = 0; d < 64; ++d) Vst[d][r] = zb;
        }
    }
#if HAVE_ASYNC
    wait_async0();
#endif
    __syncthreads();

    const int khalf = (lane >> 4) * 8;
    const int kq    = (lane >> 4) * 16;
    const int ml    = lane & 15;

    // ---- scores: Sc[64][224] = (Q . K^T) / sqrt(64).  56 tile-jobs, 7 per wave.
    for (int j = 0; j < 7; ++j) {
        int job = wave * 7 + j;
        int mt = job / 14, nt = job % 14;
        v8f acc = {};
        #pragma unroll
        for (int kk = 0; kk < 2; ++kk) {
            const bf16* pa = &Qs[mt * 16 + ml][kk * 32];
            AFrag ua;
            ua.h[0] = *(const v8bf*)(pa + khalf);
            ua.h[1] = *(const v8bf*)(pa + 16 + khalf);
            AFrag ub;
            ub.h[0] = *(const v8bf*)&Ks[(nt * 16 + ml) * 64 + kk * 32 + khalf];
            ub.h[1] = *(const v8bf*)&Ks[(nt * 16 + ml) * 64 + kk * 32 + 16 + khalf];
            v16bf bv = ub.v;
            acc = wmma_bf16f32(ua.v, bv, acc);
        }
        int mr = mt * 16 + ((lane >> 4) * 8);
        int n  = nt * 16 + ml;
        #pragma unroll
        for (int r = 0; r < 8; ++r) Sc[mr + r][n] = acc[r] * 0.125f;
    }
    __syncthreads();

    // ---- softmax + relative-position bias (one thread per query row)
    bf16* Ps = &Ks[0];
    if (t < 64) {
        int row = t, qy = row >> 3, qx = row & 7;
        float mx = -3.0e38f;
        for (int j = 0; j < 196; ++j) {
            int ky = j / 14, kx = j % 14;
            int pos = (ky - qy + 7) * 14 + (kx - qx + 7);
            float s = Sc[row][j] + rel[pos * 8 + head];
            Sc[row][j] = s;
            mx = fmaxf(mx, s);
        }
        float sum = 0.0f;
        for (int j = 0; j < 196; ++j) {
            float e = __expf(Sc[row][j] - mx);
            Sc[row][j] = e;
            sum += e;
        }
        float inv = 1.0f / sum;
        for (int j = 0; j < 196; ++j) Ps[row * 224 + j] = f2bf(Sc[row][j] * inv);
        bf16 zb = f2bf(0.0f);
        for (int j = 196; j < 224; ++j) Ps[row * 224 + j] = zb;
    }
    __syncthreads();

    // ---- out = P[64x224] . V[224x64].  16 tile-jobs, 2 per wave, 7 K-steps.
    #pragma unroll
    for (int j2 = 0; j2 < 2; ++j2) {
        int job = wave * 2 + j2;
        int mt = job >> 2, nt = job & 3;
        v8f acc = {};
        for (int kk = 0; kk < 7; ++kk) {
            const bf16* pa = Ps + (mt * 16 + ml) * 224 + kk * 32;
            AFrag ua;
            ua.h[0] = *(const v8bf*)(pa + khalf);
            ua.h[1] = *(const v8bf*)(pa + 16 + khalf);
            AFrag ub;
            ub.h[0] = *(const v8bf*)&Vst[nt * 16 + ml][kk * 32 + khalf];
            ub.h[1] = *(const v8bf*)&Vst[nt * 16 + ml][kk * 32 + 16 + khalf];
            v16bf bv = ub.v;
            acc = wmma_bf16f32(ua.v, bv, acc);
        }
        int mr = mt * 16 + ((lane >> 4) * 8);
        int n  = nt * 16 + ml;
        #pragma unroll
        for (int r = 0; r < 8; ++r) {
            size_t grow = ((size_t)(b * 64 + l)) * 64 + mr + r;
            Outp[grow * 512 + head * 64 + n] = f2bf(acc[r]);
        }
    }
}

// ---------------------------------------------------------------------------
// Kernel 4: output GEMM + bias + un-window scatter, fp32 out.
// ---------------------------------------------------------------------------
__global__ __launch_bounds__(256) void k_gemm_out(const bf16* __restrict__ Ain,
                                                  const bf16* __restrict__ Wt,
                                                  const float* __restrict__ bias,
                                                  float* __restrict__ Out) {
    __shared__ __align__(32) bf16 As[128][32];
    __shared__ __align__(32) bf16 Bs[128][32];

    const int m0 = blockIdx.x * 128;
    const int n0 = blockIdx.y * 128;
    const int t = threadIdx.x, lane = t & 31, wave = t >> 5;
    const int wm = (wave >> 2) * 64;
    const int wn = (wave & 3) * 32;
    const int lr = t >> 1, lseg = (t & 1) * 16;

    v8f acc[4][2] = {};
    const int khalf = (lane >> 4) * 8;
    const int kq    = (lane >> 4) * 16;
    const int ml    = lane & 15;

    for (int k0 = 0; k0 < 512; k0 += 32) {
        __syncthreads();
        stage_tiles(&Ain[(size_t)(m0 + lr) * 512 + k0 + lseg],
                    &Wt [(size_t)(n0 + lr) * 512 + k0 + lseg],
                    &As[lr][lseg], &Bs[lr][lseg], k0 + 32 < 512);
        __syncthreads();
        #pragma unroll
        for (int mt = 0; mt < 4; ++mt) {
            const bf16* pa = &As[wm + mt * 16 + ml][0];
            AFrag ua;
            ua.h[0] = *(const v8bf*)(pa + khalf);
            ua.h[1] = *(const v8bf*)(pa + 16 + khalf);
            #pragma unroll
            for (int nt = 0; nt < 2; ++nt) {
                AFrag ub;
                ub.h[0] = *(const v8bf*)&Bs[wn + nt * 16 + ml][khalf];
                ub.h[1] = *(const v8bf*)&Bs[wn + nt * 16 + ml][16 + khalf];
                v16bf bv = ub.v;
                acc[mt][nt] = wmma_bf16f32(ua.v, bv, acc[mt][nt]);
            }
        }
    }

    const int mrow = (lane >> 4) * 8;
    #pragma unroll
    for (int mt = 0; mt < 4; ++mt)
        #pragma unroll
        for (int nt = 0; nt < 2; ++nt) {
            int col = n0 + wn + nt * 16 + ml;
            float bv = bias[col];
            #pragma unroll
            for (int r = 0; r < 8; ++r) {
                size_t row = (size_t)(m0 + wm + mt * 16 + mrow + r);   // (b*64+l)*64+p
                int bb = (int)(row >> 12);
                int ll = (int)(row >> 6) & 63;
                int pp = (int)row & 63;
                int h = ((ll >> 3) << 3) + (pp >> 3);
                int w = ((ll & 7) << 3) + (pp & 7);
                Out[(((size_t)(bb * 64 + h)) * 64 + w) * 512 + col] = acc[mt][nt][r] + bv;
            }
        }
}

// ---------------------------------------------------------------------------
extern "C" void kernel_launch(void* const* d_in, const int* in_sizes, int n_in,
                              void* d_out, int out_size, void* d_ws, size_t ws_size,
                              hipStream_t stream) {
    (void)in_sizes; (void)n_in; (void)out_size; (void)ws_size;
    const float* x      = (const float*)d_in[0];   // [8,64,64,512]
    const float* w_qkv  = (const float*)d_in[1];   // [512,1536]
    const float* w_out  = (const float*)d_in[2];   // [512,512]
    const float* b_out  = (const float*)d_in[3];   // [512]
    const float* rel    = (const float*)d_in[4];   // [301,8]

    char* ws = (char*)d_ws;
    bf16* xb      = (bf16*)(ws);                                   //  32 MB
    bf16* wqkvT   = (bf16*)(ws + 33554432);                        // 1.5 MB [1536][512]
    bf16* woutT   = (bf16*)(ws + 33554432 + 1572864);              // 0.5 MB [512][512]
    bf16* qkvb    = (bf16*)(ws + 33554432 + 1572864 + 524288);     //  96 MB [32768][1536]
    bf16* attnout = (bf16*)(ws + 33554432 + 1572864 + 524288 + 100663296); // 32 MB
    float* out = (float*)d_out;

    k_f32_to_bf16   <<<8192, 256, 0, stream>>>(x, xb, 32768 * 512);
    k_transpose_bf16<<<(512 * 1536) / 256, 256, 0, stream>>>(w_qkv, wqkvT, 512, 1536);
    k_transpose_bf16<<<(512 * 512) / 256, 256, 0, stream>>>(w_out, woutT, 512, 512);
    k_gemm_qkv      <<<dim3(256, 12), 256, 0, stream>>>(xb, wqkvT, qkvb);
    k_halo_attn     <<<4096, 256, 0, stream>>>(qkvb, rel, attnout);
    k_gemm_out      <<<dim3(256, 4), 256, 0, stream>>>(attnout, woutT, b_out, out);
}
